// VectorQuantizer_18219251269656
// MI455X (gfx1250) — compile-verified, matrix operand order checked by reading, not run on hardware
//
#include <hip/hip_runtime.h>
#include <hip/hip_bf16.h>
#include <math.h>

typedef _Float16 half8  __attribute__((ext_vector_type(8)));
typedef _Float16 v16h   __attribute__((ext_vector_type(16)));
typedef float    v8f    __attribute__((ext_vector_type(8)));

#define VQ_N 32768
#define VQ_K 8192
#define VQ_D 256
#define VQ_BETA 0.25f
#define VQ_KSPLIT 2                 // split-K factor over the codebook
#define VQ_TILES_PER_SPLIT (VQ_K / 16 / VQ_KSPLIT)   // 256

#if defined(__has_builtin)
#if __has_builtin(__builtin_amdgcn_sched_barrier)
#define SCHED_BARRIER() __builtin_amdgcn_sched_barrier(0)
#endif
#endif
#ifndef SCHED_BARRIER
#define SCHED_BARRIER()
#endif

union FragU { half8 h[2]; v16h v; };

// ---------------------------------------------------------------------------
// Stage 1: L2-normalize rows of a [nrows x 256] fp32 matrix.
// One wave (32 lanes) per row. Writes fp16 normalized rows + 1/norm per row.
// ---------------------------------------------------------------------------
__global__ __launch_bounds__(256) void vq_normalize_rows(
    const float* __restrict__ src, _Float16* __restrict__ dst16,
    float* __restrict__ invn, int nrows)
{
    int row  = (int)((blockIdx.x * blockDim.x + threadIdx.x) >> 5);
    int lane = (int)(threadIdx.x & 31);
    if (row >= nrows) return;

    const float* p = src + (size_t)row * VQ_D;
    float vals[8];
    float s = 0.0f;
#pragma unroll
    for (int j = 0; j < 8; ++j) {
        vals[j] = p[lane + j * 32];
        s += vals[j] * vals[j];
    }
#pragma unroll
    for (int m = 16; m >= 1; m >>= 1)
        s += __shfl_xor(s, m, 32);

    float inv = 1.0f / fmaxf(sqrtf(s), 1e-12f);
#pragma unroll
    for (int j = 0; j < 8; ++j)
        dst16[(size_t)row * VQ_D + lane + j * 32] = (_Float16)(vals[j] * inv);
    if (lane == 0) invn[row] = inv;
}

// ---------------------------------------------------------------------------
// Stage 2: fused GEMM + running argmax, split-K over the codebook.
// C[n][k] = zn[n] . wn[k]  (both L2-normalized; argmin(dist) == argmax(dot)).
// Grid: (N/64, KSPLIT). One wave = 16 z-rows x (K/KSPLIT) codebook columns,
// 8 x v_wmma_f32_16x16x32_f16 per 16-column tile (D=256 = 8*32).
//
// Fragment layout (wave32, per ISA 7.12.2):
//   A (16x32 f16): lane L: row = L&15; halfs 0..7  -> K = (L>>4)*8 + 0..7
//                                      halfs 8..15 -> K = 16 + (L>>4)*8 + 0..7
//   B (32x16 f16): identical pattern with lane = output column (codebook row).
//   C (16x16 f32): elem v: lanes 0-15 -> (M=v,   N=lane)
//                          lanes16-31 -> (M=v+8, N=lane-16)
// ---------------------------------------------------------------------------
__global__ __launch_bounds__(128) void vq_argmax(
    const _Float16* __restrict__ A16,   // [N][256] normalized z, fp16
    const _Float16* __restrict__ W16,   // [K][256] normalized codebook, fp16
    int*   __restrict__ kidx2,          // [KSPLIT][N] best index per split
    float* __restrict__ kval2)          // [KSPLIT][N] best dot per split
{
    const int lane = (int)(threadIdx.x & 31);
    const int wave = (int)(threadIdx.x >> 5);
    const int r    = lane & 15;
    const int hi   = lane >> 4;
    const int row0 = ((int)blockIdx.x * 4 + wave) * 16;
    const int half = (int)blockIdx.y;                    // K split index
    const int t0   = half * VQ_TILES_PER_SPLIT;

    // Resident A fragments: 8 chunks of K=32 covering D=256.
    v16h a[8];
    {
        const _Float16* ap = A16 + (size_t)(row0 + r) * VQ_D + hi * 8;
#pragma unroll
        for (int c = 0; c < 8; ++c) {
            FragU u;
            u.h[0] = *(const half8*)(ap + c * 32);
            u.h[1] = *(const half8*)(ap + c * 32 + 16);
            a[c] = u.v;
        }
    }

    float bv[8];
    int   bn[8];
#pragma unroll
    for (int v = 0; v < 8; ++v) { bv[v] = -3.0e38f; bn[v] = t0 * 16; }

    for (int t = t0; t < t0 + VQ_TILES_PER_SPLIT; ++t) {
        const _Float16* bp = W16 + (size_t)(t * 16 + r) * VQ_D + hi * 8;

        // Speculative prefetch of the next tile's rows (safe OOB per ISA 10.5).
        __builtin_prefetch(bp + 16 * VQ_D, 0, 3);

        v16h b[8];
#pragma unroll
        for (int c = 0; c < 8; ++c) {
            FragU u;
            u.h[0] = *(const half8*)(bp + c * 32);
            u.h[1] = *(const half8*)(bp + c * 32 + 16);
            b[c] = u.v;
        }
        // Force all 16 B loads into flight before the first WMMA so the
        // waits become progressive instead of load->wait0->wmma lockstep.
        SCHED_BARRIER();

        v8f acc = {};
#pragma unroll
        for (int c = 0; c < 8; ++c)
            acc = __builtin_amdgcn_wmma_f32_16x16x32_f16(
                false, a[c], false, b[c], (short)0, acc, false, false);

        const int n = t * 16 + r;   // codebook column this lane owns
#pragma unroll
        for (int v = 0; v < 8; ++v) {
            bool p = acc[v] > bv[v];          // branchless: keep EXEC uniform
            bv[v] = p ? acc[v] : bv[v];
            bn[v] = p ? n      : bn[v];
        }
    }

    // Reduce (max, argmin-on-tie) across the 16 lanes of each half-wave.
#pragma unroll
    for (int v = 0; v < 8; ++v) {
#pragma unroll
        for (int m = 8; m >= 1; m >>= 1) {
            float ov = __shfl_xor(bv[v], m, 32);
            int   oi = __shfl_xor(bn[v], m, 32);
            bool  p  = (ov > bv[v]) || (ov == bv[v] && oi < bn[v]);
            bv[v] = p ? ov : bv[v];
            bn[v] = p ? oi : bn[v];
        }
    }
    if (r == 0) {
#pragma unroll
        for (int v = 0; v < 8; ++v) {
            int row = row0 + hi * 8 + v;
            kidx2[(size_t)half * VQ_N + row] = bn[v];
            kval2[(size_t)half * VQ_N + row] = bv[v];
        }
    }
}

// ---------------------------------------------------------------------------
// Stage 4: merge split-K candidates, gather z_q = normalize(codebook)[k],
// straight-through output, fp32 commitment-loss partials, utilization bitmap.
// One wave per row.
// ---------------------------------------------------------------------------
__global__ __launch_bounds__(256) void vq_gather(
    const float* __restrict__ z_e, const float* __restrict__ codebook,
    const float* __restrict__ invz, const float* __restrict__ invw,
    const int* __restrict__ kidx2, const float* __restrict__ kval2,
    float* __restrict__ out_zq,       // [N][256]
    float* __restrict__ out_k,        // [N]
    unsigned int* __restrict__ bitmap,
    float* __restrict__ acc)
{
    int row  = (int)((blockIdx.x * blockDim.x + threadIdx.x) >> 5);
    int lane = (int)(threadIdx.x & 31);
    if (row >= VQ_N) return;

    // Merge the KSPLIT candidates (indices in split 0 are always smaller,
    // so tie-break toward split 0 == argmin-first semantics).
    int   k  = kidx2[row];
    float kv = kval2[row];
#pragma unroll
    for (int hf = 1; hf < VQ_KSPLIT; ++hf) {
        int   ko = kidx2[(size_t)hf * VQ_N + row];
        float vo = kval2[(size_t)hf * VQ_N + row];
        bool  p  = vo > kv;
        k  = p ? ko : k;
        kv = p ? vo : kv;
    }

    float iz = invz[row];
    float iw = invw[k];
    const float* zp = z_e      + (size_t)row * VQ_D;
    const float* wp = codebook + (size_t)k   * VQ_D;

    float s = 0.0f;
#pragma unroll
    for (int j = 0; j < 8; ++j) {
        int d   = lane + j * 32;
        float z = zp[d] * iz;
        float q = wp[d] * iw;
        out_zq[(size_t)row * VQ_D + d] = z + (q - z);  // straight-through
        float diff = z - q;
        s += diff * diff;
    }
#pragma unroll
    for (int m = 16; m >= 1; m >>= 1)
        s += __shfl_xor(s, m, 32);

    if (lane == 0) {
        out_k[row] = (float)k;
        atomicAdd(acc, s);
        atomicOr(&bitmap[k >> 5], 1u << (k & 31));
    }
}

// ---------------------------------------------------------------------------
// Stage 5: finalize scalars. Single block of 256 threads: popcount the
// K/32 = 256 bitmap words, write vq_loss and utilization.
// ---------------------------------------------------------------------------
__global__ __launch_bounds__(256) void vq_finalize(
    const unsigned int* __restrict__ bitmap, const float* __restrict__ acc,
    float* __restrict__ out_loss, float* __restrict__ out_util)
{
    __shared__ int cnt[256];
    int t = (int)threadIdx.x;
    cnt[t] = __popc(bitmap[t]);
    __syncthreads();
    for (int s = 128; s > 0; s >>= 1) {
        if (t < s) cnt[t] += cnt[t + s];
        __syncthreads();
    }
    if (t == 0) {
        *out_loss = VQ_BETA * (*acc) / (float)((double)VQ_N * (double)VQ_D);
        *out_util = (float)cnt[0] / (float)VQ_K;
    }
}

// ---------------------------------------------------------------------------
extern "C" void kernel_launch(void* const* d_in, const int* in_sizes, int n_in,
                              void* d_out, int out_size, void* d_ws, size_t ws_size,
                              hipStream_t stream)
{
    (void)in_sizes; (void)n_in; (void)out_size; (void)ws_size;

    const float* z_e      = (const float*)d_in[0];   // [N][D]
    const float* codebook = (const float*)d_in[1];   // [K][D]

    float* out      = (float*)d_out;
    float* out_zq   = out;                            // N*D
    float* out_k    = out + (size_t)VQ_N * VQ_D;      // N
    float* out_loss = out_k + VQ_N;                   // 1
    float* out_util = out_loss + 1;                   // 1

    // Workspace layout (bytes), all 16B-aligned.
    char* ws = (char*)d_ws;
    size_t o = 0;
    _Float16*     A16    = (_Float16*)(ws + o); o += (size_t)VQ_N * VQ_D * 2;          // 16 MiB
    _Float16*     W16    = (_Float16*)(ws + o); o += (size_t)VQ_K * VQ_D * 2;          //  4 MiB
    float*        invz   = (float*)(ws + o);    o += (size_t)VQ_N * 4;
    float*        invw   = (float*)(ws + o);    o += (size_t)VQ_K * 4;
    int*          kidx2  = (int*)(ws + o);      o += (size_t)VQ_KSPLIT * VQ_N * 4;
    float*        kval2  = (float*)(ws + o);    o += (size_t)VQ_KSPLIT * VQ_N * 4;
    unsigned int* bitmap = (unsigned int*)(ws + o); o += (size_t)(VQ_K / 32) * 4;      // 1 KiB
    float*        acc    = (float*)(ws + o);

    // Stage 0: zero bitmap + loss accumulator (graph-capturable memset).
    hipMemsetAsync(bitmap, 0, (VQ_K / 32) * 4 + 16, stream);

    // Stage 1: normalize z and codebook rows -> fp16 + 1/norm.
    vq_normalize_rows<<<VQ_N / 8, 256, 0, stream>>>(z_e, A16, invz, VQ_N);
    vq_normalize_rows<<<VQ_K / 8, 256, 0, stream>>>(codebook, W16, invw, VQ_K);

    // Stage 2: fused WMMA GEMM + argmax, split-K for occupancy.
    vq_argmax<<<dim3(VQ_N / 64, VQ_KSPLIT), 128, 0, stream>>>(A16, W16, kidx2, kval2);

    // Stage 4: merge + gather + straight-through + loss + utilization bits.
    vq_gather<<<VQ_N / 8, 256, 0, stream>>>(z_e, codebook, invz, invw,
                                            kidx2, kval2, out_zq, out_k,
                                            bitmap, acc);

    // Stage 5: scalars.
    vq_finalize<<<1, 256, 0, stream>>>(bitmap, acc, out_loss, out_util);
}
